// SentenceEncoder_14353780703452
// MI455X (gfx1250) — matrix operand order repaired and checked by reading, not compiled, not run on hardware
//
#include <hip/hip_runtime.h>
#include <hip/hip_bf16.h>
#include <stdint.h>

#define VOCAB 32000
#define EMB 512
#define HID 1024
#define BATCH 128
#define MAXLEN 64
#define GATES (3 * HID) // 3072
#define LPAD 1032       // padded LDS row (1024 + 8 bf16) -> 2064B stride, conflict-free b128

typedef __bf16 v16bf __attribute__((ext_vector_type(16)));
typedef float v8f __attribute__((ext_vector_type(8)));

union Frag {
  v16bf v;
  uint4 q[2];
  unsigned short u[16];
};

__device__ __forceinline__ unsigned short f2bf(float f) {
  unsigned int x = __float_as_uint(f);
  x += 0x7FFFu + ((x >> 16) & 1u); // round-to-nearest-even
  return (unsigned short)(x >> 16);
}
__device__ __forceinline__ float bf2f(unsigned short u) {
  return __uint_as_float(((unsigned int)u) << 16);
}
__device__ __forceinline__ float sigm(float x) {
  return 1.0f / (1.0f + __expf(-x));
}
__device__ __forceinline__ float tanh_fast(float x) {
  float t = __expf(-2.0f * fabsf(x));
  float r = (1.0f - t) / (1.0f + t);
  return copysignf(r, x);
}
__device__ __forceinline__ v8f vzero8() {
  v8f z;
#pragma unroll
  for (int i = 0; i < 8; ++i) z[i] = 0.0f;
  return z;
}

// ---------- f32 -> bf16 convert ----------
__global__ void k_f2bf(const float* __restrict__ src, unsigned short* __restrict__ dst, int n) {
  int i = blockIdx.x * blockDim.x + threadIdx.x;
  if (i < n) dst[i] = f2bf(src[i]);
}

// ---------- gather embedding rows to bf16: embA[r][e], r = b*64+t ----------
__global__ void k_gather(const int* __restrict__ sent, const float* __restrict__ emb,
                         unsigned short* __restrict__ dst) {
  int r = blockIdx.x; // 0..8191
  int tok = sent[r];
  const float* s = emb + (size_t)tok * EMB;
  unsigned short* d = dst + (size_t)r * EMB;
  for (int e = threadIdx.x; e < EMB; e += blockDim.x) d[e] = f2bf(s[e]);
}

// ---------- x_gates GEMM: [8192 x 3072] = embA[8192 x 512] * w_ih^T, +b_ih ----------
// Output transposed bf16: xgt[t][g][b]  (t<64, g<3072, b<128)
__global__ void k_xgates(const unsigned short* __restrict__ embA,
                         const unsigned short* __restrict__ wih_bf,
                         const float* __restrict__ bih,
                         unsigned short* __restrict__ xgt) {
  int wv = (blockIdx.x * blockDim.x + threadIdx.x) >> 5; // global wave id
  int lane = threadIdx.x & 31;
  int hsel = lane >> 4;
  int ln = lane & 15;
  int mtile = wv / 48; // 0..511
  int ngrp = wv % 48;  // 4 n-tiles each
  int nt0 = ngrp * 4;

  v8f acc[4];
#pragma unroll
  for (int i = 0; i < 4; ++i) acc[i] = vzero8();

  const unsigned short* arow = embA + (size_t)(mtile * 16 + ln) * EMB;
  const unsigned short* brow[4];
#pragma unroll
  for (int i = 0; i < 4; ++i) brow[i] = wih_bf + (size_t)((nt0 + i) * 16 + ln) * EMB;

  for (int kt = 0; kt < EMB / 32; ++kt) {
    // batch ALL loads of this k-step first, then all WMMAs (enables partial waits)
    Frag a, b[4];
    a.q[0] = *(const uint4*)(arow + kt * 32 + 8 * hsel);
    a.q[1] = *(const uint4*)(arow + kt * 32 + 16 + 8 * hsel);
    int k0 = kt * 32 + 8 * hsel;
#pragma unroll
    for (int i = 0; i < 4; ++i) {
      b[i].q[0] = *(const uint4*)(brow[i] + k0);
      b[i].q[1] = *(const uint4*)(brow[i] + k0 + 16);
    }
#pragma unroll
    for (int i = 0; i < 4; ++i) {
      acc[i] = __builtin_amdgcn_wmma_f32_16x16x32_bf16(false, a.v, false, b[i].v,
                                                       (short)0, acc[i], false, false);
    }
  }

#pragma unroll
  for (int i = 0; i < 4; ++i) {
    int g = (nt0 + i) * 16 + ln;
    float bias = bih[g];
#pragma unroll
    for (int v = 0; v < 8; ++v) {
      int r = mtile * 16 + 8 * hsel + v; // flat (b,t) row
      int t = r & 63;
      int bb = r >> 6;
      xgt[((size_t)t * GATES + g) * BATCH + bb] = f2bf(acc[i][v] + bias);
    }
  }
}

// ---------- persistent GRU recurrence: one block (32 waves) per 16-row batch chunk ----------
// wave w owns hidden tiles jt = w*2 + q (q=0..1); per tile computes r/z/n gate columns.
__global__ void __launch_bounds__(1024, 1)
k_gru(const unsigned short* __restrict__ whh_bf, const float* __restrict__ bhh,
      const unsigned short* __restrict__ xgt, const int* __restrict__ slen,
      float* __restrict__ hfinal) {
  __shared__ unsigned short lh[16 * LPAD]; // bf16 h, padded rows

  int tid = threadIdx.x;
  int w = tid >> 5; // 0..31
  int lane = tid & 31;
  int hsel = lane >> 4;
  int ln = lane & 15;
  int b0 = blockIdx.x * 16;

  int lenm1[8];
#pragma unroll
  for (int v = 0; v < 8; ++v) lenm1[v] = slen[b0 + 8 * hsel + v] - 1;

  for (int i = tid; i < 16 * LPAD; i += 1024) lh[i] = 0;

  v8f hreg[2];
#pragma unroll
  for (int q = 0; q < 2; ++q) hreg[q] = vzero8();

  // per-wave column ownership: hidden tiles jt = w*2+q, columns j = jt*16+ln
  float bh_r[2], bh_z[2], bh_n[2];
  const unsigned short* wr[6]; // [q*3 + {r,z,n}]
#pragma unroll
  for (int q = 0; q < 2; ++q) {
    int j = (w * 2 + q) * 16 + ln;
    bh_r[q] = bhh[j];
    bh_z[q] = bhh[HID + j];
    bh_n[q] = bhh[2 * HID + j];
    wr[q * 3 + 0] = whh_bf + (size_t)j * HID;
    wr[q * 3 + 1] = whh_bf + (size_t)(HID + j) * HID;
    wr[q * 3 + 2] = whh_bf + (size_t)(2 * HID + j) * HID;
  }
  __syncthreads();

  for (int t = 0; t < MAXLEN; ++t) {
    v8f acc[6]; // [q*3 + {r,z,n}]
#pragma unroll
    for (int i = 0; i < 6; ++i) acc[i] = vzero8();

    // ---- phase 1: hg = h @ whh^T  (reads lh) ----
    for (int kt = 0; kt < HID / 32; ++kt) {
      // batch ALL loads first, then all WMMAs
      Frag a, b[6];
      const unsigned short* ab = lh + ln * LPAD + kt * 32;
      a.q[0] = *(const uint4*)(ab + 8 * hsel);
      a.q[1] = *(const uint4*)(ab + 16 + 8 * hsel);
      int k0 = kt * 32 + 8 * hsel;
#pragma unroll
      for (int i = 0; i < 6; ++i) {
        b[i].q[0] = *(const uint4*)(wr[i] + k0);
        b[i].q[1] = *(const uint4*)(wr[i] + k0 + 16);
      }
#pragma unroll
      for (int i = 0; i < 6; ++i) {
        acc[i] = __builtin_amdgcn_wmma_f32_16x16x32_bf16(false, a.v, false, b[i].v,
                                                         (short)0, acc[i], false, false);
      }
    }
    __syncthreads(); // all reads of lh done

    // ---- phase 2: gates + h update (writes lh) ----
    const unsigned short* xbase = xgt + (size_t)t * GATES * BATCH;
#pragma unroll
    for (int q = 0; q < 2; ++q) {
      int j = (w * 2 + q) * 16 + ln;
      Frag xr, xz, xn;
      xr.q[0] = *(const uint4*)(xbase + (size_t)j * BATCH + b0 + 8 * hsel);
      xz.q[0] = *(const uint4*)(xbase + (size_t)(HID + j) * BATCH + b0 + 8 * hsel);
      xn.q[0] = *(const uint4*)(xbase + (size_t)(2 * HID + j) * BATCH + b0 + 8 * hsel);
#pragma unroll
      for (int v = 0; v < 8; ++v) {
        float hr = acc[q * 3 + 0][v] + bh_r[q];
        float hz = acc[q * 3 + 1][v] + bh_z[q];
        float hn = acc[q * 3 + 2][v] + bh_n[q];
        float rg = sigm(bf2f(xr.u[v]) + hr);
        float zg = sigm(bf2f(xz.u[v]) + hz);
        float ng = tanh_fast(bf2f(xn.u[v]) + rg * hn);
        float hnew = (1.0f - zg) * ng + zg * hreg[q][v];
        hreg[q][v] = hnew;
        int m = 8 * hsel + v;
        lh[m * LPAD + j] = f2bf(hnew);
        if (t == lenm1[v]) hfinal[(size_t)(b0 + m) * HID + j] = hnew;
      }
    }
    __syncthreads(); // lh fully updated before next step's reads
  }
}

// ---------- L2 normalize rows ----------
__global__ void k_norm(const float* __restrict__ hfinal, float* __restrict__ out) {
  __shared__ float red[256];
  int b = blockIdx.x;
  int tid = threadIdx.x;
  const float* row = hfinal + (size_t)b * HID;
  float s = 0.0f;
  for (int i = tid; i < HID; i += 256) {
    float v = row[i];
    s += v * v;
  }
  red[tid] = s;
  __syncthreads();
  for (int off = 128; off > 0; off >>= 1) {
    if (tid < off) red[tid] += red[tid + off];
    __syncthreads();
  }
  float nrm = sqrtf(red[0]);
  for (int i = tid; i < HID; i += 256) out[(size_t)b * HID + i] = row[i] / nrm;
}

extern "C" void kernel_launch(void* const* d_in, const int* in_sizes, int n_in,
                              void* d_out, int out_size, void* d_ws, size_t ws_size,
                              hipStream_t stream) {
  (void)in_sizes; (void)n_in; (void)out_size; (void)ws_size;
  const int* sent = (const int*)d_in[0];    // [128,64]
  const int* slen = (const int*)d_in[1];    // [128]
  const float* emb = (const float*)d_in[2]; // [32000,512]
  const float* wih = (const float*)d_in[3]; // [3072,512]
  const float* whh = (const float*)d_in[4]; // [3072,1024]
  const float* bih = (const float*)d_in[5]; // [3072]
  const float* bhh = (const float*)d_in[6]; // [3072]
  float* out = (float*)d_out;               // [128,1024] f32

  char* ws = (char*)d_ws;
  unsigned short* wih_bf = (unsigned short*)(ws);            //  3,145,728 B
  unsigned short* whh_bf = (unsigned short*)(ws + 3145728);  //  6,291,456 B
  unsigned short* embA   = (unsigned short*)(ws + 9437184);  //  8,388,608 B
  unsigned short* xgt    = (unsigned short*)(ws + 17825792); // 50,331,648 B
  float* hfinal          = (float*)(ws + 68157440);          //    524,288 B

  k_f2bf<<<(3072 * 512 + 255) / 256, 256, 0, stream>>>(wih, wih_bf, 3072 * 512);
  k_f2bf<<<(3072 * 1024 + 255) / 256, 256, 0, stream>>>(whh, whh_bf, 3072 * 1024);
  k_gather<<<BATCH * MAXLEN, 128, 0, stream>>>(sent, emb, embA);
  k_xgates<<<3072, 256, 0, stream>>>(embA, wih_bf, bih, xgt); // 512 mtiles * 48 ngroups waves
  k_gru<<<BATCH / 16, 1024, 0, stream>>>(whh_bf, bhh, xgt, slen, hfinal);
  k_norm<<<BATCH, 256, 0, stream>>>(hfinal, out);
}
